// TransformerBlock_39118562132313
// MI455X (gfx1250) — matrix operand order repaired and checked by reading, not compiled, not run on hardware
//
#include <hip/hip_runtime.h>
#include <math.h>

// ---------------------------------------------------------------------------
// Types for CDNA5 WMMA (wave32, 16x16x32 bf16 -> f32)
// ---------------------------------------------------------------------------
typedef __bf16 v16bf __attribute__((ext_vector_type(16)));
typedef __bf16 bf16x8 __attribute__((ext_vector_type(8)));
typedef float  v8f   __attribute__((ext_vector_type(8)));

#define DIM      384
#define NUM_H    8
#define HD       48
#define HDP      64      // padded head dim for K-dim of QK^T
#define HIDDEN   1024
#define NTOK     2048
#define BATCH    4
#define ROWS     (BATCH * NTOK)   // 8192
#define SCALE_F  0.14433756729740643f  // 48^-0.5
#define EPS_F    1e-5f

// ---------------------------------------------------------------------------
// Fragment loaders.
// A (16x32 bf16, MxK): lane m = lane&15; halves 0-7 = K [8*hi .. 8*hi+7],
//                      halves 8-15 = K [16+8*hi .. 23+8*hi]   (ISA A table)
// B (32x16 bf16, KxN): lane n = lane&15; halves 0-7 = K [8*hi .. 8*hi+7],
//                      halves 8-15 = K [16+8*hi .. 23+8*hi]   (same order as A)
// Row pointers must point at 32 contiguous K values for this lane's m / n.
// ---------------------------------------------------------------------------
__device__ inline v16bf load_fragA(const __bf16* row, int hi) {
    bf16x8 lo = *(const bf16x8*)(row + 8 * hi);
    bf16x8 up = *(const bf16x8*)(row + 16 + 8 * hi);
    v16bf r;
#pragma unroll
    for (int i = 0; i < 8; ++i) { r[i] = lo[i]; r[i + 8] = up[i]; }
    return r;
}
__device__ inline v16bf load_fragB(const __bf16* row, int hi) {
    bf16x8 lo = *(const bf16x8*)(row + 8 * hi);
    bf16x8 up = *(const bf16x8*)(row + 16 + 8 * hi);
    v16bf r;
#pragma unroll
    for (int i = 0; i < 8; ++i) { r[i] = lo[i]; r[i + 8] = up[i]; }
    return r;
}

// ---------------------------------------------------------------------------
// Weight cast fp32 -> bf16 (same [O][K] layout; O-row-major == WMMA B layout)
// ---------------------------------------------------------------------------
__global__ void wcast_kernel(const float* __restrict__ W, __bf16* __restrict__ Wb,
                             int n) {
    int idx = blockIdx.x * blockDim.x + threadIdx.x;
    if (idx >= n) return;
    Wb[idx] = (__bf16)W[idx];
}

// ---------------------------------------------------------------------------
// RMSNorm (row of 384) -> bf16
// ---------------------------------------------------------------------------
__global__ __launch_bounds__(128)
void rmsnorm_bf16_kernel(const float* __restrict__ x, const float* __restrict__ w,
                         __bf16* __restrict__ out) {
    int row = blockIdx.x, t = threadIdx.x;
    const float* xr = x + (size_t)row * DIM;
    float s = 0.f;
    for (int c = t; c < DIM; c += 128) { float v = xr[c]; s += v * v; }
    __shared__ float red[128];
    red[t] = s; __syncthreads();
    for (int st = 64; st > 0; st >>= 1) {
        if (t < st) red[t] += red[t + st];
        __syncthreads();
    }
    float rs = rsqrtf(red[0] * (1.f / DIM) + EPS_F);
    for (int c = t; c < DIM; c += 128)
        out[(size_t)row * DIM + c] = (__bf16)(xr[c] * rs * w[c]);
}

// ---------------------------------------------------------------------------
// Tiled bf16 GEMM: C[M x N] = A[M x K] * W^T, with W given as [N][K] bf16
// (both operands K-contiguous => both LDS tile fills are 128-bit vectorized).
// Block: 256 threads = 8 waves (4x2); block tile 128x64; wave tile 32x32.
// M % 128 == 0, N % 64 == 0, K % 32 == 0.
// ---------------------------------------------------------------------------
__global__ __launch_bounds__(256)
void gemm_bf16_kernel(const __bf16* __restrict__ A, const __bf16* __restrict__ W,
                      float* __restrict__ C, int M, int N, int K) {
    __shared__ __align__(16) __bf16 sA[128][40];  // [m][k]
    __shared__ __align__(16) __bf16 sB[64][40];   // [n][k]
    const int t    = threadIdx.x;
    const int lane = t & 31;
    const int wave = t >> 5;
    const int wr = wave >> 1, wc = wave & 1;      // 4x2 wave grid
    const int hi = lane >> 4, ln = lane & 15;
    const int m0 = blockIdx.y * 128;
    const int n0 = blockIdx.x * 64;

    v8f acc[2][2] = {};

    for (int kt = 0; kt < K; kt += 32) {
        // A tile: 128 rows x 32 cols, 512 x 16B chunks, 2 per thread
#pragma unroll
        for (int j = 0; j < 2; ++j) {
            int idx = t + 256 * j;
            int r = idx >> 2, c = (idx & 3) * 8;
            *(bf16x8*)&sA[r][c] = *(const bf16x8*)(A + (size_t)(m0 + r) * K + kt + c);
        }
        // B tile: 64 n-rows x 32 k, 256 x 16B chunks, 1 per thread (no transpose)
        {
            int r = t >> 2, c = (t & 3) * 8;
            *(bf16x8*)&sB[r][c] = *(const bf16x8*)(W + (size_t)(n0 + r) * K + kt + c);
        }
        __syncthreads();

        v16bf bfrag[2];
#pragma unroll
        for (int j2 = 0; j2 < 2; ++j2)
            bfrag[j2] = load_fragB(&sB[wc * 32 + j2 * 16 + ln][0], hi);
#pragma unroll
        for (int i = 0; i < 2; ++i) {
            v16bf afrag = load_fragA(&sA[wr * 32 + i * 16 + ln][0], hi);
#pragma unroll
            for (int j2 = 0; j2 < 2; ++j2)
                acc[i][j2] = __builtin_amdgcn_wmma_f32_16x16x32_bf16(
                    false, afrag, false, bfrag[j2], (short)0, acc[i][j2], false, false);
        }
        __syncthreads();
    }
#pragma unroll
    for (int i = 0; i < 2; ++i)
#pragma unroll
        for (int j2 = 0; j2 < 2; ++j2)
#pragma unroll
            for (int r = 0; r < 8; ++r)
                C[(size_t)(m0 + wr * 32 + i * 16 + r + 8 * hi) * N +
                  n0 + wc * 32 + j2 * 16 + ln] = acc[i][j2][r];
}

// ---------------------------------------------------------------------------
// Assemble q/k/v from fused QKV output.
// q,k: [B*H][N][64] bf16 (48 real dims + zero pad); q folded with SCALE.
// vT : [B*H][48][N] bf16 (transposed so PV B-fragments are contiguous).
// ---------------------------------------------------------------------------
__global__ __launch_bounds__(128)
void assemble_qkv_kernel(const float* __restrict__ qkv, const float* __restrict__ pos,
                         __bf16* __restrict__ qb, __bf16* __restrict__ kb,
                         __bf16* __restrict__ vt) {
    int bn = blockIdx.x;                 // over B*N
    int b = bn / NTOK, n = bn - b * NTOK;
    const float* base = qkv + (size_t)bn * (3 * DIM);
    const float* pe   = pos + (size_t)bn * HD;
    for (int i = threadIdx.x; i < NUM_H * HDP; i += 128) {
        int h = i >> 6, d = i & 63;
        size_t oidx = ((size_t)(b * NUM_H + h) * NTOK + n) * HDP + d;
        if (d < HD) {
            float p = pe[d];
            qb[oidx] = (__bf16)((base[0 * DIM + h * HD + d] + p) * SCALE_F);
            kb[oidx] = (__bf16)(base[1 * DIM + h * HD + d] + p);
        } else {
            qb[oidx] = (__bf16)0.0f;
            kb[oidx] = (__bf16)0.0f;
        }
    }
    for (int i = threadIdx.x; i < DIM; i += 128) {
        int h = i / HD, d = i - h * HD;
        vt[((size_t)(b * NUM_H + h) * HD + d) * NTOK + n] = (__bf16)base[2 * DIM + i];
    }
}

// ---------------------------------------------------------------------------
// Flash attention. grid = (B*H, N/64); 128 threads = 4 waves, one 16-row
// Q-tile per wave. Online softmax with 16-lane shfl_xor reductions.
// Writes O directly as bf16 in [B*N][384] layout (ready for proj GEMM).
// ---------------------------------------------------------------------------
__global__ __launch_bounds__(128)
void attn_kernel(const __bf16* __restrict__ Q, const __bf16* __restrict__ Kb,
                 const __bf16* __restrict__ Vt, const float* __restrict__ mask,
                 __bf16* __restrict__ Obf) {
    const int bh = blockIdx.x;
    const int b = bh / NUM_H, h = bh - b * NUM_H;
    const int t = threadIdx.x, lane = t & 31, wave = t >> 5;
    const int hi = lane >> 4, ln = lane & 15;
    const int row0 = blockIdx.y * 64 + wave * 16;

    __shared__ __align__(16) __bf16 sP[4][16][40];   // per-wave P scratch
    __bf16 (*myP)[40] = sP[wave];

    const __bf16* qp = Q + ((size_t)bh * NTOK + row0) * HDP;
    v16bf qa0 = load_fragA(qp + (size_t)ln * HDP, hi);        // K dims 0..31
    v16bf qa1 = load_fragA(qp + (size_t)ln * HDP + 32, hi);   // K dims 32..63

    v8f o[3] = {};
    float mrun[8], lsum[8];
#pragma unroll
    for (int r = 0; r < 8; ++r) { mrun[r] = -3.0e38f; lsum[r] = 0.f; }

    const float*  mbase = mask + (size_t)b * NTOK * NTOK;
    const __bf16* kbase = Kb + (size_t)bh * NTOK * HDP;
    const __bf16* vbase = Vt + (size_t)bh * HD * NTOK;

    for (int mc = 0; mc < NTOK; mc += 32) {
        // ---- scores S[16x32] = Q (16x64) * K^T (64x32) ----
        v8f s0 = {}, s1 = {};
        {
            const __bf16* k0 = kbase + (size_t)(mc + ln) * HDP;        // keys mc..mc+15
            v16bf kb0a = load_fragB(k0, hi);
            v16bf kb0b = load_fragB(k0 + 32, hi);
            s0 = __builtin_amdgcn_wmma_f32_16x16x32_bf16(false, qa0, false, kb0a, (short)0, s0, false, false);
            s0 = __builtin_amdgcn_wmma_f32_16x16x32_bf16(false, qa1, false, kb0b, (short)0, s0, false, false);
            const __bf16* k1 = kbase + (size_t)(mc + 16 + ln) * HDP;   // keys mc+16..mc+31
            v16bf kb1a = load_fragB(k1, hi);
            v16bf kb1b = load_fragB(k1 + 32, hi);
            s1 = __builtin_amdgcn_wmma_f32_16x16x32_bf16(false, qa0, false, kb1a, (short)0, s1, false, false);
            s1 = __builtin_amdgcn_wmma_f32_16x16x32_bf16(false, qa1, false, kb1b, (short)0, s1, false, false);
        }
        // ---- + mask, online softmax ----
#pragma unroll
        for (int r = 0; r < 8; ++r) {
            const float* mr = mbase + (size_t)(row0 + r + 8 * hi) * NTOK + mc;
            float v0 = s0[r] + mr[ln];
            float v1 = s1[r] + mr[16 + ln];
            float mx = fmaxf(v0, v1);
#pragma unroll
            for (int off = 1; off < 16; off <<= 1) mx = fmaxf(mx, __shfl_xor(mx, off, 32));
            float mnew = fmaxf(mrun[r], mx);
            float corr = __expf(mrun[r] - mnew);
            mrun[r] = mnew;
            float p0 = __expf(v0 - mnew);
            float p1 = __expf(v1 - mnew);
            float ps = p0 + p1;
#pragma unroll
            for (int off = 1; off < 16; off <<= 1) ps += __shfl_xor(ps, off, 32);
            lsum[r] = lsum[r] * corr + ps;
            o[0][r] *= corr; o[1][r] *= corr; o[2][r] *= corr;
            myP[r + 8 * hi][ln]      = (__bf16)p0;   // C-layout -> LDS row-major
            myP[r + 8 * hi][16 + ln] = (__bf16)p1;
        }
        // ---- O += P (16x32) * V (32x48) ----  (LDS is in-order per wave)
        v16bf pa = load_fragA(&myP[ln][0], hi);
#pragma unroll
        for (int nd = 0; nd < 3; ++nd) {
            v16bf vb = load_fragB(vbase + (size_t)(nd * 16 + ln) * NTOK + mc, hi);
            o[nd] = __builtin_amdgcn_wmma_f32_16x16x32_bf16(false, pa, false, vb, (short)0, o[nd], false, false);
        }
    }
    // ---- normalize + write bf16 O in [B*N][384] ----
#pragma unroll
    for (int r = 0; r < 8; ++r) {
        float inv = 1.f / lsum[r];
        size_t rowi = (size_t)(b * NTOK + row0 + r + 8 * hi) * DIM + h * HD;
#pragma unroll
        for (int nd = 0; nd < 3; ++nd)
            Obf[rowi + nd * 16 + ln] = (__bf16)(o[nd][r] * inv);
    }
}

// ---------------------------------------------------------------------------
// h = x + proj + b_proj ; hf = rmsnorm(h)*norm2_w (bf16)
// ---------------------------------------------------------------------------
__global__ __launch_bounds__(128)
void residual_rms2_kernel(const float* __restrict__ x, const float* __restrict__ proj,
                          const float* __restrict__ bproj, const float* __restrict__ w,
                          float* __restrict__ hbuf, __bf16* __restrict__ hf) {
    int row = blockIdx.x, t = threadIdx.x;
    const float* xr = x + (size_t)row * DIM;
    const float* pr = proj + (size_t)row * DIM;
    float s = 0.f;
    for (int c = t; c < DIM; c += 128) {
        float v = xr[c] + pr[c] + bproj[c];
        hbuf[(size_t)row * DIM + c] = v;
        s += v * v;
    }
    __shared__ float red[128];
    red[t] = s; __syncthreads();
    for (int st = 64; st > 0; st >>= 1) {
        if (t < st) red[t] += red[t + st];
        __syncthreads();
    }
    float rs = rsqrtf(red[0] * (1.f / DIM) + EPS_F);
    for (int c = t; c < DIM; c += 128)
        hf[(size_t)row * DIM + c] = (__bf16)(hbuf[(size_t)row * DIM + c] * rs * w[c]);
}

// ---------------------------------------------------------------------------
// g = silu(a1) * a3  (bf16)
// ---------------------------------------------------------------------------
__global__ void swiglu_kernel(const float* __restrict__ a1, const float* __restrict__ a3,
                              __bf16* __restrict__ g, int n) {
    int idx = blockIdx.x * blockDim.x + threadIdx.x;
    if (idx >= n) return;
    float v = a1[idx];
    float s = v / (1.f + __expf(-v));
    g[idx] = (__bf16)(s * a3[idx]);
}

// ---------------------------------------------------------------------------
// out = keep ? h + mlp : 0   (num_points may be int32 or int64 on device)
// ---------------------------------------------------------------------------
__global__ void final_kernel(const float* __restrict__ hbuf, const float* __restrict__ mlp,
                             const int* __restrict__ npraw, float* __restrict__ out, int n) {
    int idx = blockIdx.x * blockDim.x + threadIdx.x;
    if (idx >= n) return;
    int bn = idx / DIM;
    int b = bn / NTOK, tok = bn - b * NTOK;
    // heuristic: int64 little-endian has zero high words (values < 2048)
    bool is64 = (npraw[1] == 0) && (npraw[3] == 0);
    int np = is64 ? npraw[2 * b] : npraw[b];
    out[idx] = (tok < np) ? (hbuf[idx] + mlp[idx]) : 0.f;
}

// ---------------------------------------------------------------------------
// Launch
// ---------------------------------------------------------------------------
extern "C" void kernel_launch(void* const* d_in, const int* in_sizes, int n_in,
                              void* d_out, int out_size, void* d_ws, size_t ws_size,
                              hipStream_t stream) {
    const float* x       = (const float*)d_in[0];
    const float* pos     = (const float*)d_in[1];
    const float* mask    = (const float*)d_in[2];
    const int*   np      = (const int*)d_in[3];
    const float* norm1_w = (const float*)d_in[4];
    const float* norm2_w = (const float*)d_in[5];
    const float* w_qkv   = (const float*)d_in[6];
    const float* w_proj  = (const float*)d_in[7];
    const float* b_proj  = (const float*)d_in[8];
    const float* w1      = (const float*)d_in[9];
    const float* w2      = (const float*)d_in[10];
    const float* w3      = (const float*)d_in[11];
    float* out = (float*)d_out;

    // ---- workspace layout (256B aligned slabs, aggressive reuse) ----
    char* ws = (char*)d_ws;
    size_t off = 0;
    auto alloc = [&](size_t bytes) {
        char* p = ws + off;
        off += (bytes + 255) & ~(size_t)255;
        return p;
    };
    __bf16* xa     = (__bf16*)alloc((size_t)ROWS * DIM * 2);            // 6.3 MB
    __bf16* wqkvB  = (__bf16*)alloc((size_t)3 * DIM * DIM * 2);         // 0.9 MB
    __bf16* wprojB = (__bf16*)alloc((size_t)DIM * DIM * 2);             // 0.3 MB
    __bf16* w1B    = (__bf16*)alloc((size_t)HIDDEN * DIM * 2);          // 0.8 MB
    __bf16* w3B    = (__bf16*)alloc((size_t)HIDDEN * DIM * 2);          // 0.8 MB
    __bf16* w2B    = (__bf16*)alloc((size_t)DIM * HIDDEN * 2);          // 0.8 MB
    float*  big1   = (float*)alloc((size_t)ROWS * 3 * DIM * 4);         // 37.7 MB : qkv, then a1
    float*  big2   = (float*)alloc((size_t)ROWS * HIDDEN * 4);          // 33.6 MB : proj, then a3
    __bf16* qb     = (__bf16*)alloc((size_t)BATCH * NUM_H * NTOK * HDP * 2); // 8.4 MB
    __bf16* kb     = (__bf16*)alloc((size_t)BATCH * NUM_H * NTOK * HDP * 2); // 8.4 MB (qb+kb reused as g)
    __bf16* vt     = (__bf16*)alloc((size_t)BATCH * NUM_H * HD * NTOK * 2);  // 6.3 MB
    __bf16* obf    = (__bf16*)alloc((size_t)ROWS * DIM * 2);            // 6.3 MB (vt+obf reused as mlp)
    float*  hbuf   = (float*)alloc((size_t)ROWS * DIM * 4);             // 12.6 MB
    __bf16* hf     = (__bf16*)alloc((size_t)ROWS * DIM * 2);            // 6.3 MB
    float*  qkv    = big1;
    float*  proj   = big2;
    float*  a1     = big1;
    float*  a3     = big2;
    __bf16* g      = qb;            // reuse (16.78 MB needed == qb+kb)
    float*  mlp    = (float*)vt;    // reuse (12.58 MB needed == vt+obf)

    // ---- weight casts (straight cast; [O][K] layout is WMMA-B-ready) ----
    auto wc = [&](const float* W, __bf16* Wb, int n) {
        wcast_kernel<<<(n + 255) / 256, 256, 0, stream>>>(W, Wb, n);
    };
    wc(w_qkv,  wqkvB,  3 * DIM * DIM);
    wc(w_proj, wprojB, DIM * DIM);
    wc(w1,     w1B,    HIDDEN * DIM);
    wc(w3,     w3B,    HIDDEN * DIM);
    wc(w2,     w2B,    DIM * HIDDEN);

    // ---- 1) rmsnorm1 ----
    rmsnorm_bf16_kernel<<<ROWS, 128, 0, stream>>>(x, norm1_w, xa);

    // ---- 2) QKV GEMM: [8192x384] x [1152x384]^T ----
    gemm_bf16_kernel<<<dim3((3 * DIM) / 64, ROWS / 128), 256, 0, stream>>>(
        xa, wqkvB, qkv, ROWS, 3 * DIM, DIM);

    // ---- 3) assemble q/k/v (+pos, scale folded into q) ----
    assemble_qkv_kernel<<<ROWS, 128, 0, stream>>>(qkv, pos, qb, kb, vt);

    // ---- 4) attention ----
    attn_kernel<<<dim3(BATCH * NUM_H, NTOK / 64), 128, 0, stream>>>(qb, kb, vt, mask, obf);

    // ---- 5) proj GEMM: [8192x384] x [384x384]^T ----
    gemm_bf16_kernel<<<dim3(DIM / 64, ROWS / 128), 256, 0, stream>>>(
        obf, wprojB, proj, ROWS, DIM, DIM);

    // ---- 6) residual + rmsnorm2 ----
    residual_rms2_kernel<<<ROWS, 128, 0, stream>>>(x, proj, b_proj, norm2_w, hbuf, hf);

    // ---- 7) MLP up GEMMs ----
    gemm_bf16_kernel<<<dim3(HIDDEN / 64, ROWS / 128), 256, 0, stream>>>(
        hf, w1B, a1, ROWS, HIDDEN, DIM);
    gemm_bf16_kernel<<<dim3(HIDDEN / 64, ROWS / 128), 256, 0, stream>>>(
        hf, w3B, a3, ROWS, HIDDEN, DIM);

    // ---- 8) swiglu ----
    {
        int n = ROWS * HIDDEN;
        swiglu_kernel<<<(n + 255) / 256, 256, 0, stream>>>(a1, a3, g, n);
    }

    // ---- 9) MLP down GEMM: [8192x1024] x [384x1024]^T ----
    gemm_bf16_kernel<<<dim3(DIM / 64, ROWS / 128), 256, 0, stream>>>(
        g, w2B, mlp, ROWS, DIM, HIDDEN);

    // ---- 10) final residual + num_points mask ----
    {
        int n = ROWS * DIM;
        final_kernel<<<(n + 255) / 256, 256, 0, stream>>>(hbuf, mlp, np, out, n);
    }
}
